// GCN_Reg_38354057954042
// MI455X (gfx1250) — matrix operand order repaired and checked by reading, not compiled, not run on hardware
//
#include <hip/hip_runtime.h>
#include <hip/hip_bf16.h>
#include <stdint.h>

#define NN     16384
#define IN_F   128
#define HID    256
#define KSTAGE 64          // K elements staged per double-buffer step (2 wmma k-blocks)
#define MBLK   32

typedef __attribute__((ext_vector_type(16))) __bf16 v16bf;
typedef __attribute__((ext_vector_type(8)))  float  v8f;

static __device__ __forceinline__ unsigned short f2bf(float f) {
    unsigned u = __builtin_bit_cast(unsigned, f);
    unsigned r = u + 0x7FFFu + ((u >> 16) & 1u);   // round-to-nearest-even
    return (unsigned short)(r >> 16);
}

// two floats -> packed bf16x2 (round-half-up) : 2x v_add + 1x v_perm_b32
static __device__ __forceinline__ unsigned pack_bf16x2(float a, float b) {
#if __has_builtin(__builtin_amdgcn_cvt_pk_bf16_f32)
    typedef __attribute__((ext_vector_type(2))) __bf16 v2bf;
    v2bf p = __builtin_amdgcn_cvt_pk_bf16_f32(a, b);
    return __builtin_bit_cast(unsigned, p);
#else
    unsigned ua = __builtin_bit_cast(unsigned, a) + 0x8000u;
    unsigned ub = __builtin_bit_cast(unsigned, b) + 0x8000u;
    // bytes {0,1} = ua[31:16], bytes {2,3} = ub[31:16]
    return __builtin_amdgcn_perm(ub, ua, 0x07060302u);
#endif
}

static __device__ __forceinline__ uint2 pack_bf16x4(float4 v) {
    uint2 pk;
    pk.x = pack_bf16x2(v.x, v.y);
    pk.y = pack_bf16x2(v.z, v.w);
    return pk;
}

static __device__ __forceinline__ void wait_async0() {
#if __has_builtin(__builtin_amdgcn_s_wait_asynccnt)
    __builtin_amdgcn_s_wait_asynccnt(0);
#else
    asm volatile("s_wait_asynccnt 0" ::: "memory");
#endif
}

static __device__ __forceinline__ unsigned lds_off(const void* p) {
    return (unsigned)(uintptr_t)p;   // low 32 bits of generic LDS addr = LDS offset
}

// ---------------------------------------------------------------------------
// Kernel 1: xw = x @ W1, stored bf16 in WMMA B-fragment order:
//   xwp[ ((kblk*16 + ntile)*32 + lane)*16 + elem ]
//   kblk=k/32, kk=k%32, lane = (n%16) + 16*(kk/16), elem = kk%16
// ---------------------------------------------------------------------------
__global__ __launch_bounds__(256) void k_xw_pack(const float* __restrict__ x,
                                                 const float* __restrict__ W1,
                                                 unsigned short* __restrict__ xwp) {
    __shared__ float xs[16 * IN_F];
    const int tid = threadIdx.x;
    const int r0  = blockIdx.x * 16;

    for (int i = tid; i < 16 * IN_F; i += 256) {
        int r = i >> 7, c = i & 127;
        xs[i] = x[(size_t)(r0 + r) * IN_F + c];
    }
    __syncthreads();

    float acc[16];
#pragma unroll
    for (int r = 0; r < 16; ++r) acc[r] = 0.f;

    const int n = tid;                      // 256 threads == HID columns
    for (int k = 0; k < IN_F; ++k) {
        float w = W1[k * HID + n];
#pragma unroll
        for (int r = 0; r < 16; ++r) acc[r] += xs[r * IN_F + k] * w;
    }

    const int nt = n >> 4, nn = n & 15;
#pragma unroll
    for (int r = 0; r < 16; ++r) {
        int kg   = r0 + r;
        int kblk = kg >> 5, kk = kg & 31;
        int lane = nn + ((kk >> 4) << 4);
        int e    = kk & 15;
        xwp[(((size_t)kblk * 16 + nt) * 32 + lane) * 16 + e] = f2bf(acc[r]);
    }
}

// ---------------------------------------------------------------------------
// Kernel 2: fused  hw = relu(adj @ xw + b1) @ W2   (h never stored)
// Block: 256 threads = 8 waves. Tile: 32 rows x 256 cols, K staged 64 at a time.
// ---------------------------------------------------------------------------
__global__ __launch_bounds__(256) void k_layer1(const float* __restrict__ adj,
                                                const unsigned short* __restrict__ xwp,
                                                const float* __restrict__ b1,
                                                const float* __restrict__ W2,
                                                float* __restrict__ hw) {
    __shared__ unsigned short ldsB[2][2 * 16 * 32 * 16]; // 2 x 32 KB bf16 B fragments
    __shared__ unsigned short ldsA[2][MBLK * KSTAGE];    // 2 x 4 KB  bf16 A tile
    __shared__ float ldsHW[MBLK];

    const int tid     = threadIdx.x;
    const int wave    = tid >> 5;
    const int lane    = tid & 31;
    const int row_blk = blockIdx.x * MBLK;
    const int mt      = wave >> 2;                     // row half of the 32-row tile
    const int nq      = wave & 3;                      // 64-col quarter
    const int half    = lane >> 4;
    const int ml      = mt * 16 + (lane & 15);         // row within 32-row tile

    const unsigned ldsB_base = lds_off(&ldsB[0][0]);
    const int NSTG = NN / KSTAGE;                      // 256

    // A-staging mapping: 8 threads per row; two float4 chunks at col and col+32
    const int arow = tid >> 3;
    const int acol = (tid & 7) * 4;
    const size_t a_row_base = (size_t)(row_blk + arow) * NN + acol;

    v8f acc[4] = {};
    float4 av0, av1;

    // ---- prologue: fill buffer 0 (stage 0 => K 0..63) ----
    {
        const char* src = (const char*)xwp;
#pragma unroll
        for (int i = 0; i < 8; ++i) {
            uint64_t g = (uint64_t)(uintptr_t)(src + i * 4096 + tid * 16);
            unsigned d = ldsB_base + i * 4096 + tid * 16;
            asm volatile("global_load_async_to_lds_b128 %0, %1, off"
                         :: "v"(d), "v"(g) : "memory");
        }
        av0 = *(const float4*)(adj + a_row_base);
        av1 = *(const float4*)(adj + a_row_base + 32);
        *(uint2*)&ldsA[0][arow * KSTAGE + acol]      = pack_bf16x4(av0);
        *(uint2*)&ldsA[0][arow * KSTAGE + acol + 32] = pack_bf16x4(av1);
        wait_async0();
    }
    __syncthreads();

    for (int kb = 0; kb < NSTG; ++kb) {
        const int cur = kb & 1;
        const int nxt = cur ^ 1;
        const bool more = (kb + 1) < NSTG;

        if (more) {
            const char* src = (const char*)xwp + (size_t)(kb + 1) * 32768;
            const unsigned dstb = ldsB_base + (unsigned)nxt * 32768u;
#pragma unroll
            for (int i = 0; i < 8; ++i) {
                uint64_t g = (uint64_t)(uintptr_t)(src + i * 4096 + tid * 16);
                unsigned d = dstb + i * 4096 + tid * 16;
                asm volatile("global_load_async_to_lds_b128 %0, %1, off"
                             :: "v"(d), "v"(g) : "memory");
            }
            av0 = *(const float4*)(adj + a_row_base + (size_t)(kb + 1) * KSTAGE);
            av1 = *(const float4*)(adj + a_row_base + (size_t)(kb + 1) * KSTAGE + 32);
        }

        // ---- compute on cur: 2 sub-blocks of K=32, 4 WMMAs each ----
#pragma unroll
        for (int s = 0; s < 2; ++s) {
            union { v16bf v; uint4 q[2]; } afr;
            union { v16bf v; uint4 q[2]; } bfr[4];
            afr.q[0] = *(const uint4*)&ldsA[cur][ml * KSTAGE + s * 32 + half * 16];
            afr.q[1] = *(const uint4*)&ldsA[cur][ml * KSTAGE + s * 32 + half * 16 + 8];
#pragma unroll
            for (int j = 0; j < 4; ++j) {
                const int nt = nq * 4 + j;
                const uint4* bp =
                    (const uint4*)&ldsB[cur][s * 8192 + (nt * 32 + lane) * 16];
                bfr[j].q[0] = bp[0];
                bfr[j].q[1] = bp[1];
            }
#pragma unroll
            for (int j = 0; j < 4; ++j) {
                acc[j] = __builtin_amdgcn_wmma_f32_16x16x32_bf16(
                    false, afr.v, false, bfr[j].v, (short)0, acc[j], false, false);
            }
        }

        if (more) {
            *(uint2*)&ldsA[nxt][arow * KSTAGE + acol]      = pack_bf16x4(av0);
            *(uint2*)&ldsA[nxt][arow * KSTAGE + acol + 32] = pack_bf16x4(av1);
            wait_async0();
        }
        __syncthreads();
    }

    // ---- epilogue: h = relu(acc + b1);  p[r] = sum_n h * W2[n] ----
    float p[8];
#pragma unroll
    for (int r = 0; r < 8; ++r) p[r] = 0.f;
#pragma unroll
    for (int j = 0; j < 4; ++j) {
        const int n = nq * 64 + j * 16 + (lane & 15);
        const float bb = b1[n];
        const float w2 = W2[n];
#pragma unroll
        for (int r = 0; r < 8; ++r) {
            float h = fmaxf(acc[j][r] + bb, 0.f);
            p[r] += h * w2;
        }
    }
    // reduce over the 16 lanes (n dimension) sharing each row
#pragma unroll
    for (int off = 1; off < 16; off <<= 1) {
#pragma unroll
        for (int r = 0; r < 8; ++r) p[r] += __shfl_xor(p[r], off, 32);
    }

    if (tid < MBLK) ldsHW[tid] = 0.f;
    __syncthreads();
    if ((lane & 15) < 8) {
        const int rr = lane & 15;
        const int m  = mt * 16 + half * 8 + rr;
        atomicAdd(&ldsHW[m], p[rr]);                  // ds_add_f32, 4 waves per row
    }
    __syncthreads();
    if (tid < MBLK) hw[row_blk + tid] = ldsHW[tid];
}

// ---------------------------------------------------------------------------
// Kernel 3: out = relu(adj @ hw + b2)   — memory-bound fp32 GEMV
// ---------------------------------------------------------------------------
__global__ __launch_bounds__(256) void k_layer2(const float* __restrict__ adj,
                                                const float* __restrict__ hw,
                                                const float* __restrict__ b2,
                                                float* __restrict__ out) {
    __shared__ float red[256];
    const int row = blockIdx.x;
    const int tid = threadIdx.x;
    const float4* arow = (const float4*)(adj + (size_t)row * NN);
    const float4* h4   = (const float4*)hw;
    float s = 0.f;
#pragma unroll 4
    for (int i = tid; i < NN / 4; i += 256) {
        float4 a = arow[i], h = h4[i];
        s += a.x * h.x + a.y * h.y + a.z * h.z + a.w * h.w;
    }
    red[tid] = s;
    __syncthreads();
    for (int st = 128; st > 0; st >>= 1) {
        if (tid < st) red[tid] += red[tid + st];
        __syncthreads();
    }
    if (tid == 0) out[row] = fmaxf(red[0] + b2[0], 0.f);
}

// ---------------------------------------------------------------------------
extern "C" void kernel_launch(void* const* d_in, const int* in_sizes, int n_in,
                              void* d_out, int out_size, void* d_ws, size_t ws_size,
                              hipStream_t stream) {
    const float* x   = (const float*)d_in[0];
    const float* adj = (const float*)d_in[1];
    const float* W1  = (const float*)d_in[2];
    const float* b1  = (const float*)d_in[3];
    const float* W2  = (const float*)d_in[4];
    const float* b2  = (const float*)d_in[5];
    float* out = (float*)d_out;

    unsigned short* xwp = (unsigned short*)d_ws;                  // 8 MB bf16 packed xw
    float* hw = (float*)((char*)d_ws + (size_t)NN * HID * 2);     // 64 KB

    k_xw_pack<<<NN / 16, 256, 0, stream>>>(x, W1, xwp);
    k_layer1 <<<NN / MBLK, 256, 0, stream>>>(adj, xwp, b1, W2, hw);
    k_layer2 <<<NN, 256, 0, stream>>>(adj, hw, b2, out);
}
